// TraceableBaseLMStep_66709432041916
// MI455X (gfx1250) — matrix operand order repaired and checked by reading, not compiled, not run on hardware
//
#include <hip/hip_runtime.h>
#include <hip/hip_bf16.h>

#define L_    24
#define HID_  1024
#define NH_   16
#define HD_   64
#define S_    1536
#define S1_   1537
#define FF_   2816
#define FSQ_  32
#define EPS_  1e-6f
#define SCALE_ 0.2857738033247042f   /* 1.4 / sqrt(24) */

typedef __attribute__((ext_vector_type(16))) _Float16 v16h;
typedef __attribute__((ext_vector_type(8)))  float    v8f;

// ---------------------------------------------------------------------------
// Wave-level multi-matrix WMMA GEMV tile:
//   c[m] holds out_m[n0..n0+15] = sum_k x[k] * W_m[k*Nd + n]
// x is replicated across the 16 M rows of A (16x32 f16 layout, ISA 7.12.2);
// each B tile is a 32(K) x 16(N) block of W_m, loaded f32 -> cvt f16.
// Sharing A across NMAT matrices amortizes activation fetch and gives NMAT
// independent WMMA chains per k-chunk for latency hiding.
// Result row M=0 of D: c[m][0] on lanes 0..15 (N = lane).
// ---------------------------------------------------------------------------
template <int NMAT>
__device__ __forceinline__ void gemv_tile_n(const _Float16* xh,
                                            const float* const (&W)[NMAT],
                                            int Kd, int Nd, int n0,
                                            v8f (&c)[NMAT]) {
  const int lane = threadIdx.x & 31;
  const int col  = lane & 15;
  const int hi   = lane >> 4;     // lane group 0/1
  const int ab   = hi * 8;        // A k-offset within chunk
  const int kb2  = hi * 8;        // B k-offset within chunk
  const float* wp[NMAT];
#pragma unroll
  for (int m = 0; m < NMAT; ++m) wp[m] = W[m] + (size_t)n0 + col;
  for (int kb = 0; kb < Kd; kb += 32) {
    v16h a;
#pragma unroll
    for (int j = 0; j < 8; ++j) {
      a[j]     = xh[kb + ab + j];        // K = ab + j       (lanes<16: 0..7,  else 8..15)
      a[j + 8] = xh[kb + 16 + ab + j];   // K = 16 + ab + j  (16..23 / 24..31)
    }
#pragma unroll
    for (int m = 0; m < NMAT; ++m) {
      v16h b;
#pragma unroll
      for (int jj = 0; jj < 16; ++jj) {
        b[jj] = (_Float16)wp[m][(size_t)(kb + kb2 + (jj < 8 ? jj : jj + 8)) * Nd];
      }
      c[m] = __builtin_amdgcn_wmma_f32_16x16x32_f16(false, a, false, b,
                                                    (short)0, c[m], false, false);
    }
  }
}

// Block-level RMSNorm into an LDS f16 vector (recomputed per block; ~8KB
// redundant reads per block vs >10MB of streamed weights: free).
__device__ __forceinline__ void block_rms(const float* __restrict__ h,
                                          const float* __restrict__ w,
                                          _Float16* xh, float* red) {
  const int tid = threadIdx.x, nt = blockDim.x;
  float s = 0.f;
  for (int k = tid; k < HID_; k += nt) { float v = h[k]; s += v * v; }
  red[tid] = s; __syncthreads();
  for (int off = nt >> 1; off > 0; off >>= 1) {
    if (tid < off) red[tid] += red[tid + off];
    __syncthreads();
  }
  float rs = rsqrtf(red[0] * (1.0f / HID_) + EPS_);
  for (int k = tid; k < HID_; k += nt) xh[k] = (_Float16)(h[k] * rs * w[k]);
  __syncthreads();
}

// ---------------------------------------------------------------------------
__global__ void k_copy_cache(const float* __restrict__ Kin,
                             const float* __restrict__ Vin,
                             float* __restrict__ outK, float* __restrict__ outV) {
  int b = blockIdx.x;
  const float* src; float* dst;
  if (b < L_ * NH_) { src = Kin + (size_t)b * (S_ * HD_); dst = outK + (size_t)b * (S1_ * HD_); }
  else { b -= L_ * NH_; src = Vin + (size_t)b * (S_ * HD_); dst = outV + (size_t)b * (S1_ * HD_); }
  for (int i = threadIdx.x; i < S_ * HD_; i += blockDim.x) dst[i] = src[i];
}

__global__ void k_init(const float* __restrict__ e, float* __restrict__ h) {
  for (int k = threadIdx.x; k < HID_; k += blockDim.x) h[k] = e[k];
}

// Fused RMS + QKV projection: 16 blocks x 128 threads = 64 wave-tiles, each
// wave produces the same 16-column tile of q, k and v (A shared, 3 WMMA chains).
__global__ void k_qkv(const float* __restrict__ h, const float* __restrict__ ln1,
                      const float* __restrict__ Wq, const float* __restrict__ Wk,
                      const float* __restrict__ Wv,
                      float* __restrict__ qo, float* __restrict__ ko,
                      float* __restrict__ vo, int layer) {
  __shared__ _Float16 xh[HID_];
  __shared__ float red[128];
  block_rms(h, ln1 + (size_t)layer * HID_, xh, red);
  const int wid = threadIdx.x >> 5, lane = threadIdx.x & 31;
  const int n0 = (blockIdx.x * 4 + wid) << 4;
  const size_t off = (size_t)layer * HID_ * HID_;
  const float* W[3] = { Wq + off, Wk + off, Wv + off };
  v8f c[3] = {};
  gemv_tile_n<3>(xh, W, HID_, HID_, n0, c);
  if (lane < 16) {
    qo[n0 + lane] = c[0][0];
    ko[n0 + lane] = c[1][0];
    vo[n0 + lane] = c[2][0];
  }
}

// One block per head: RoPE(q,k), write s=1536 cache rows, scores, softmax, A*V.
__global__ void k_attn(const float* __restrict__ Kin, const float* __restrict__ Vin,
                       const float* __restrict__ qws, const float* __restrict__ kws,
                       const float* __restrict__ vws, const int* __restrict__ pos,
                       float* __restrict__ outK, float* __restrict__ outV,
                       float* __restrict__ ows, _Float16* __restrict__ oh, int layer) {
  __shared__ __align__(16) float qr[HD_];
  __shared__ __align__(16) float kr[HD_];
  __shared__ float vn[HD_], rawq[HD_], rawk[HD_];
  __shared__ float sc[S1_];
  __shared__ float red[256];
  const int hh = blockIdx.x, tid = threadIdx.x;
  const size_t lh = (size_t)layer * NH_ + hh;

  if (tid < HD_) {
    rawq[tid] = qws[hh * HD_ + tid];
    rawk[tid] = kws[hh * HD_ + tid];
    vn[tid]   = vws[hh * HD_ + tid];
  }
  __syncthreads();
  if (tid < HD_) {
    const float pf = (float)pos[0];
    const int j = tid & 31;
    const float rinv = __expf(-(float)j * 0.28782313662425575f); // ln(1e4)/32
    const float ang = pf * rinv;
    const float cs = cosf(ang), sn = sinf(ang);
    const float rotq = (tid < 32) ? -rawq[tid + 32] : rawq[tid - 32];
    const float rotk = (tid < 32) ? -rawk[tid + 32] : rawk[tid - 32];
    qr[tid] = rawq[tid] * cs + rotq * sn;
    kr[tid] = rawk[tid] * cs + rotk * sn;
    outK[lh * (size_t)(S1_ * HD_) + (size_t)S_ * HD_ + tid] = kr[tid];
    outV[lh * (size_t)(S1_ * HD_) + (size_t)S_ * HD_ + tid] = vn[tid];
  }
  __syncthreads();
  // scores: hot loop is pure global_load_b128 (no generic/flat selection)
  for (int s = tid; s < S_; s += 256) {
    const float4* row = (const float4*)(Kin + (lh * S_ + s) * (size_t)HD_);
    float acc = 0.f;
#pragma unroll
    for (int j = 0; j < 16; ++j) {
      float4 kk = row[j];
      acc += kk.x * qr[4 * j] + kk.y * qr[4 * j + 1] +
             kk.z * qr[4 * j + 2] + kk.w * qr[4 * j + 3];
    }
    sc[s] = acc * 0.125f;
  }
  if (tid == 0) {                // s = 1536 row from LDS
    float acc = 0.f;
    for (int j = 0; j < HD_; ++j) acc += kr[j] * qr[j];
    sc[S_] = acc * 0.125f;
  }
  __syncthreads();
  // max
  float mx = -3.4e38f;
  for (int s = tid; s < S1_; s += 256) mx = fmaxf(mx, sc[s]);
  red[tid] = mx; __syncthreads();
  for (int off = 128; off > 0; off >>= 1) {
    if (tid < off) red[tid] = fmaxf(red[tid], red[tid + off]);
    __syncthreads();
  }
  mx = red[0]; __syncthreads();
  // exp + sum
  float sm = 0.f;
  for (int s = tid; s < S1_; s += 256) { float e = __expf(sc[s] - mx); sc[s] = e; sm += e; }
  red[tid] = sm; __syncthreads();
  for (int off = 128; off > 0; off >>= 1) {
    if (tid < off) red[tid] += red[tid + off];
    __syncthreads();
  }
  const float invs = 1.f / red[0];
  __syncthreads();
  // A*V  (4 s-groups x 64 dims; coalesced along d)
  const int d = tid & 63, g = tid >> 6;
  float acc = 0.f;
  for (int s = g; s < S_; s += 4) acc += sc[s] * Vin[(lh * S_ + s) * (size_t)HD_ + d];
  if (g == 0) acc += sc[S_] * vn[d];
  red[tid] = acc; __syncthreads();
  if (tid < HD_) {
    float ov = (red[tid] + red[64 + tid] + red[128 + tid] + red[192 + tid]) * invs;
    ows[hh * HD_ + tid] = ov;
    oh[hh * HD_ + tid]  = (_Float16)ov;
  }
}

__global__ void k_oproj(const _Float16* __restrict__ oh, const float* __restrict__ Wo,
                        float* __restrict__ h, int layer) {
  const int wid = threadIdx.x >> 5, lane = threadIdx.x & 31;
  const int n0 = (blockIdx.x * 4 + wid) << 4;
  const float* W[1] = { Wo + (size_t)layer * HID_ * HID_ };
  v8f c[1] = {};
  gemv_tile_n<1>(oh, W, HID_, HID_, n0, c);
  if (lane < 16) h[n0 + lane] += SCALE_ * c[0][0];
}

// Fused RMS + gate/up GEMV + SiLU. 44 blocks x 128 = 176 tiles x 16 = 2816.
__global__ void k_gateup(const float* __restrict__ h, const float* __restrict__ ln2,
                         const float* __restrict__ Wg, const float* __restrict__ Wu,
                         float* __restrict__ m, _Float16* __restrict__ mh, int layer) {
  __shared__ _Float16 xh[HID_];
  __shared__ float red[128];
  block_rms(h, ln2 + (size_t)layer * HID_, xh, red);
  const int wid = threadIdx.x >> 5, lane = threadIdx.x & 31;
  const int n0 = (blockIdx.x * 4 + wid) << 4;
  const size_t off = (size_t)layer * HID_ * FF_;
  const float* W[2] = { Wg + off, Wu + off };
  v8f c[2] = {};
  gemv_tile_n<2>(xh, W, HID_, FF_, n0, c);
  if (lane < 16) {
    float g = c[0][0], u = c[1][0];
    float mv = g / (1.f + __expf(-g)) * u;
    m[n0 + lane] = mv;
    mh[n0 + lane] = (_Float16)mv;
  }
}

__global__ void k_down(const _Float16* __restrict__ mh, const float* __restrict__ Wd,
                       float* __restrict__ h, int layer) {
  const int wid = threadIdx.x >> 5, lane = threadIdx.x & 31;
  const int n0 = (blockIdx.x * 4 + wid) << 4;
  const float* W[1] = { Wd + (size_t)layer * FF_ * HID_ };
  v8f c[1] = {};
  gemv_tile_n<1>(mh, W, FF_, HID_, n0, c);
  if (lane < 16) h[n0 + lane] += SCALE_ * c[0][0];
}

__global__ void k_finalnorm(const float* __restrict__ h, const float* __restrict__ nw,
                            float* __restrict__ lm, _Float16* __restrict__ lmh) {
  __shared__ float red[256];
  const int tid = threadIdx.x;
  float s = 0.f;
  for (int k = tid; k < HID_; k += 256) { float v = h[k]; s += v * v; }
  red[tid] = s; __syncthreads();
  for (int off = 128; off > 0; off >>= 1) {
    if (tid < off) red[tid] += red[tid + off];
    __syncthreads();
  }
  float rs = rsqrtf(red[0] * (1.0f / HID_) + EPS_);
  for (int k = tid; k < HID_; k += 256) {
    float v = h[k] * rs * nw[k];
    lm[k] = v;
    lmh[k] = (_Float16)v;
  }
}

__global__ void k_fsq(const float* __restrict__ lm, const float* __restrict__ fw,
                      const float* __restrict__ fb, float* __restrict__ out) {
  __shared__ float red[256];
  const int tid = threadIdx.x, col = tid & 31, grp = tid >> 5;
  float s = 0.f;
  for (int k = grp * 128; k < grp * 128 + 128; ++k) s += lm[k] * fw[k * FSQ_ + col];
  red[tid] = s; __syncthreads();
  if (tid < 32) {
    float t = 0.f;
    for (int g = 0; g < 8; ++g) t += red[g * 32 + tid];
    float z = tanhf(t + fb[tid]);
    out[tid] = rintf(z * 3.5f) * (1.0f / 3.5f);   // STE forward value
  }
}

__global__ void k_stopproj(const _Float16* __restrict__ lmh, const float* __restrict__ Wsp,
                           const float* __restrict__ b, float* __restrict__ sp) {
  const int wid = threadIdx.x >> 5, lane = threadIdx.x & 31;
  const int n0 = (blockIdx.x * 4 + wid) << 4;
  const float* W[1] = { Wsp };
  v8f c[1] = {};
  gemv_tile_n<1>(lmh, W, HID_, HID_, n0, c);
  if (lane < 16) {
    float v = c[0][0] + b[n0 + lane];
    sp[n0 + lane] = v / (1.f + __expf(-v));
  }
}

__global__ void k_stophead(const float* __restrict__ sp, const float* __restrict__ w,
                           const float* __restrict__ b, float* __restrict__ out) {
  __shared__ float red[256];
  const int tid = threadIdx.x;
  float s = 0.f;
  for (int k = tid; k < HID_; k += 256) s += sp[k] * w[k];
  red[tid] = s; __syncthreads();
  for (int off = 128; off > 0; off >>= 1) {
    if (tid < off) red[tid] += red[tid + off];
    __syncthreads();
  }
  if (tid == 0) out[0] = red[0] + b[0];
}

// ---------------------------------------------------------------------------
extern "C" void kernel_launch(void* const* d_in, const int* in_sizes, int n_in,
                              void* d_out, int out_size, void* d_ws, size_t ws_size,
                              hipStream_t stream) {
  (void)in_sizes; (void)n_in; (void)out_size; (void)ws_size;
  const float* embed = (const float*)d_in[0];
  const int*   pos   = (const int*)d_in[1];
  const float* Kin   = (const float*)d_in[2];
  const float* Vin   = (const float*)d_in[3];
  const float* ln1   = (const float*)d_in[4];
  const float* ln2   = (const float*)d_in[5];
  const float* Wq    = (const float*)d_in[6];
  const float* Wk    = (const float*)d_in[7];
  const float* Wv    = (const float*)d_in[8];
  const float* Wo    = (const float*)d_in[9];
  const float* Wg    = (const float*)d_in[10];
  const float* Wu    = (const float*)d_in[11];
  const float* Wd    = (const float*)d_in[12];
  const float* nw    = (const float*)d_in[13];
  const float* fw    = (const float*)d_in[14];
  const float* fb    = (const float*)d_in[15];
  const float* spw   = (const float*)d_in[16];
  const float* spb   = (const float*)d_in[17];
  const float* shw   = (const float*)d_in[18];
  const float* shb   = (const float*)d_in[19];

  float* out      = (float*)d_out;
  float* lm_out   = out;
  float* fsq_out  = out + 1024;
  float* stop_out = out + 1056;
  float* outK     = out + 1057;
  float* outV     = outK + (size_t)L_ * NH_ * S1_ * HD_;

  float* ws  = (float*)d_ws;
  float* h   = ws;
  float* q_  = ws + 1024;
  float* k_  = ws + 2048;
  float* v_  = ws + 3072;
  float* o_  = ws + 4096;
  float* m_  = ws + 5120;
  float* sp_ = ws + 7936;
  _Float16* hb  = (_Float16*)(ws + 8960);
  _Float16* oh  = hb;          // 1024 halves
  _Float16* mh  = hb + 1024;   // 2816 halves
  _Float16* lmh = hb + 3840;   // 1024 halves

  k_copy_cache<<<2 * L_ * NH_, 256, 0, stream>>>(Kin, Vin, outK, outV);
  k_init<<<1, 256, 0, stream>>>(embed, h);
  for (int i = 0; i < L_; ++i) {
    k_qkv   <<<16, 128, 0, stream>>>(h, ln1, Wq, Wk, Wv, q_, k_, v_, i);
    k_attn  <<<NH_, 256, 0, stream>>>(Kin, Vin, q_, k_, v_, pos, outK, outV, o_, oh, i);
    k_oproj <<<16, 128, 0, stream>>>(oh, Wo, h, i);
    k_gateup<<<44, 128, 0, stream>>>(h, ln2, Wg, Wu, m_, mh, i);
    k_down  <<<16, 128, 0, stream>>>(mh, Wd, h, i);
  }
  k_finalnorm<<<1, 256, 0, stream>>>(h, nw, lm_out, lmh);
  k_fsq      <<<1, 256, 0, stream>>>(lm_out, fw, fb, fsq_out);
  k_stopproj <<<16, 128, 0, stream>>>(lmh, spw, spb, sp_);
  k_stophead <<<1, 256, 0, stream>>>(sp_, shw, shb, stop_out);
}
